// Matcher_592705487233
// MI455X (gfx1250) — compile-verified, matrix operand order checked by reading, not run on hardware
//
#include <hip/hip_runtime.h>
#include <hip/hip_bf16.h>

// Problem constants (from the reference)
static constexpr int Bn = 32;
static constexpr int Qn = 4096;
static constexpr int Gn = 32;
static constexpr int Cn = 128;
static constexpr float BIG = 1e30f;

typedef __attribute__((ext_vector_type(16))) _Float16 v16h;
typedef __attribute__((ext_vector_type(8)))  float    v8f;
typedef __attribute__((ext_vector_type(4)))  float    v4f;

// ---------------------------------------------------------------------------
// Kernel 1: cost-volume build via WMMA.
// class[b,q,g] = prob[b,q,label[b,g]]  ==  [Q,C] x onehot[C,G] GEMM per batch.
// Each wave computes a 16(q) x 32(g) tile: two 16x16 accumulators, K=C=128 in
// four 32-chunks -> 8 x v_wmma_f32_16x16x32_f16 per wave (fully unrolled).
// A is fetched with 16B vector loads (global_load_b128); B is a one-hot of the
// gt labels built in-register (v_cndmask_b16). Output is stored TRANSPOSED as
// costT[b][g][q] for the LSA kernel.
// ---------------------------------------------------------------------------
__global__ __launch_bounds__(256) void build_cost_wmma(
    const float* __restrict__ sem,   // [B][Q][C]
    const float* __restrict__ obj,   // [B][Q]
    const float* __restrict__ cen,   // [B][Q][G]
    const float* __restrict__ gio,   // [B][Q][G]
    const int*   __restrict__ lab,   // [B][G]
    float*       __restrict__ costT) // [B][G][Q]
{
    const int lane = threadIdx.x & 31;
    const int wave = threadIdx.x >> 5;
    const int gw   = blockIdx.x * 8 + wave;   // global wave id, 8192 total
    const int b    = gw >> 8;                 // 256 q-tiles per batch
    const int q0   = (gw & 255) << 4;         // q-tile base
    const int m    = lane & 15;               // row (A) / col (B) within tile
    const int hi   = lane >> 4;               // K-half selector

    // labels for the two g-tiles (lane -> n = m)
    const int lab0 = lab[b * Gn + m];
    const int lab1 = lab[b * Gn + 16 + m];

    // A-row pointer for this lane's q row
    const float* __restrict__ arow = sem + (size_t)(b * Qn + q0 + m) * Cn;

    // prefetch the epilogue rows while we do the GEMM
    __builtin_prefetch(cen + (size_t)(b * Qn + q0 + m) * Gn, 0, 0);
    __builtin_prefetch(gio + (size_t)(b * Qn + q0 + m) * Gn, 0, 0);

    v8f acc0 = {}; // g = 0..15
    v8f acc1 = {}; // g = 16..31

#pragma unroll
    for (int kc = 0; kc < Cn; kc += 32) {
        const int kb = kc + hi * 8;
        // A (16x32 f16), ISA 16-bit layout: h[0..7] = K kb+0..7,
        // h[8..15] = K kb+16..23. 16B vector loads (32B-aligned addresses).
        const v4f* __restrict__ ap = (const v4f*)(arow + kb);
        const v4f x0 = ap[0];  // kb + 0..3
        const v4f x1 = ap[1];  // kb + 4..7
        const v4f x2 = ap[4];  // kb + 16..19
        const v4f x3 = ap[5];  // kb + 20..23
        v16h a;
#pragma unroll
        for (int j = 0; j < 4; ++j) {
            a[j]      = (_Float16)x0[j];
            a[4 + j]  = (_Float16)x1[j];
            a[8 + j]  = (_Float16)x2[j];
            a[12 + j] = (_Float16)x3[j];
        }
        // B (32x16 f16): one-hot of gt labels, same K mapping mirrored onto N
        const int r0 = lab0 - kb;
        const int r1 = lab1 - kb;
        v16h b0v, b1v;
#pragma unroll
        for (int j = 0; j < 8; ++j) {
            b0v[j]     = (r0 == j)      ? (_Float16)1.0f : (_Float16)0.0f;
            b0v[8 + j] = (r0 == 16 + j) ? (_Float16)1.0f : (_Float16)0.0f;
            b1v[j]     = (r1 == j)      ? (_Float16)1.0f : (_Float16)0.0f;
            b1v[8 + j] = (r1 == 16 + j) ? (_Float16)1.0f : (_Float16)0.0f;
        }
        acc0 = __builtin_amdgcn_wmma_f32_16x16x32_f16(false, a, false, b0v,
                                                      (short)0, acc0, false, false);
        acc1 = __builtin_amdgcn_wmma_f32_16x16x32_f16(false, a, false, b1v,
                                                      (short)0, acc1, false, false);
    }

    // Epilogue: C/D layout -> element r of acc is (q = q0 + r + 8*hi, g = tile*16 + m)
#pragma unroll
    for (int r = 0; r < 8; ++r) {
        const int q = q0 + r + 8 * hi;
        const float o = obj[b * Qn + q];
        const size_t rowbase = (size_t)(b * Qn + q) * Gn;
        {
            const int g = m;
            const float c  = cen[rowbase + g];
            const float gv = gio[rowbase + g];
            costT[((size_t)b * Gn + g) * Qn + q] = -acc0[r] - 0.5f * o + c - 2.0f * gv;
        }
        {
            const int g = 16 + m;
            const float c  = cen[rowbase + g];
            const float gv = gio[rowbase + g];
            costT[((size_t)b * Gn + g) * Qn + q] = -acc1[r] - 0.5f * o + c - 2.0f * gv;
        }
    }
}

// ---------------------------------------------------------------------------
// Kernel 2: Jonker-Volgenant LSA, one block (256 threads) per batch.
// Big arrays (v, minv, way, p, used) live in global scratch (L2-resident on
// MI455X: 192MB L2 >> 18MB working set); LDS holds the duals u[32], the
// argmin reduction tree and broadcast scalars.
// ---------------------------------------------------------------------------
__global__ __launch_bounds__(256) void hungarian_kernel(
    const float* __restrict__ costT,   // [B][G][Q]
    const int*   __restrict__ nactual, // [B]
    float* __restrict__ v_ws,          // [B][Q+1]
    float* __restrict__ minv_ws,       // [B][Q]
    int*   __restrict__ way_ws,        // [B][Q]
    int*   __restrict__ p_ws,          // [B][Q+1]
    int*   __restrict__ used_ws,       // [B][Q+1]
    float* __restrict__ out)           // per_prop [B][Q] then mask [B][Q]
{
    const int b  = blockIdx.x;
    const int t  = threadIdx.x;
    const int NT = 256;

    __shared__ float u[Gn];
    __shared__ float red_val[256];
    __shared__ int   red_idx[256];
    __shared__ int   s_j0, s_i0, s_brk;
    __shared__ float s_delta;

    const float* __restrict__ Cb = costT + (size_t)b * Gn * Qn;
    float* v    = v_ws    + (size_t)b * (Qn + 1);
    float* minv = minv_ws + (size_t)b * Qn;
    int*   way  = way_ws  + (size_t)b * Qn;
    int*   p    = p_ws    + (size_t)b * (Qn + 1);
    int*   used = used_ws + (size_t)b * (Qn + 1);

    int ng = nactual[b];
    if (ng < 0) ng = 0;
    if (ng > Gn) ng = Gn;

    for (int j = t; j <= Qn; j += NT) { v[j] = 0.0f; p[j] = -1; }
    if (t < Gn) u[t] = 0.0f;
    __syncthreads();

    for (int i = 0; i < ng; ++i) {
        for (int j = t; j <= Qn; j += NT) used[j] = 0;
        for (int j = t; j < Qn;  j += NT) minv[j] = BIG;
        if (t == 0) { p[Qn] = i; s_j0 = Qn; }
        __syncthreads();

        while (true) {
            if (t == 0) {
                const int j0 = s_j0;
                used[j0] = 1;
                s_i0 = p[j0];
            }
            __syncthreads();
            const int   i0  = s_i0;
            const int   j0b = s_j0;
            const float ui0 = u[i0];
            const float* __restrict__ Crow = Cb + (size_t)i0 * Qn;

            // fused relaxation + per-thread argmin (lowest-index tie break)
            float bestv = BIG;
            int   besti = 0;
            for (int j = t; j < Qn; j += NT) {
                float mj;
                if (!used[j]) {
                    const float cur = Crow[j] - ui0 - v[j];
                    mj = minv[j];
                    if (cur < mj) { mj = cur; minv[j] = cur; way[j] = j0b; }
                } else {
                    mj = BIG;
                }
                if (mj < bestv) { bestv = mj; besti = j; }
            }
            red_val[t] = bestv;
            red_idx[t] = besti;
            __syncthreads();
            for (int s = NT / 2; s > 0; s >>= 1) {
                if (t < s) {
                    const float ov = red_val[t + s];
                    const int   oi = red_idx[t + s];
                    if (ov < red_val[t] || (ov == red_val[t] && oi < red_idx[t])) {
                        red_val[t] = ov; red_idx[t] = oi;
                    }
                }
                __syncthreads();
            }
            if (t == 0) { s_delta = red_val[0]; s_j0 = red_idx[0]; }
            __syncthreads();
            const float delta = s_delta;
            const int   j1    = s_j0;

            // dual updates (matched rows of used columns are distinct -> no races)
            for (int j = t; j <= Qn; j += NT) {
                if (used[j]) {
                    const int r = p[j];
                    if (r >= 0) u[r] += delta;
                    v[j] -= delta;
                } else if (j < Qn) {
                    minv[j] -= delta;
                }
            }
            __syncthreads();
            if (t == 0) s_brk = (p[j1] == -1) ? 1 : 0;
            __syncthreads();
            if (s_brk) break;
        }

        // augment along the alternating path (serial, thread 0)
        if (t == 0) {
            int j0 = s_j0;
            while (j0 != Qn) {
                const int j1 = way[j0];
                p[j0] = p[j1];
                j0 = j1;
            }
        }
        __syncthreads();
    }

    // outputs: per_prop then mask, both as float
    for (int j = t; j < Qn; j += NT) {
        const int r = p[j];
        out[(size_t)b * Qn + j]                    = (r >= 0) ? (float)r : 0.0f;
        out[(size_t)Bn * Qn + (size_t)b * Qn + j]  = (r >= 0) ? 1.0f    : 0.0f;
    }
}

// ---------------------------------------------------------------------------
extern "C" void kernel_launch(void* const* d_in, const int* in_sizes, int n_in,
                              void* d_out, int out_size, void* d_ws, size_t ws_size,
                              hipStream_t stream) {
    (void)in_sizes; (void)n_in; (void)out_size; (void)ws_size;

    const float* sem  = (const float*)d_in[0]; // [B,Q,C]
    const float* obj  = (const float*)d_in[1]; // [B,Q]
    const float* cen  = (const float*)d_in[2]; // [B,Q,G]
    const float* gio  = (const float*)d_in[3]; // [B,Q,G]
    const int*   lab  = (const int*)d_in[4];   // [B,G]
    const int*   nact = (const int*)d_in[5];   // [B]
    float* out = (float*)d_out;

    // Workspace carving (floats first, then ints)
    float* costT = (float*)d_ws;                       // B*G*Q
    size_t off = (size_t)Bn * Gn * Qn;
    float* v_ws    = costT + off;  off += (size_t)Bn * (Qn + 1);
    float* minv_ws = costT + off;  off += (size_t)Bn * Qn;
    int* iws = (int*)(costT + off);
    int* way_ws  = iws;
    int* p_ws    = way_ws + (size_t)Bn * Qn;
    int* used_ws = p_ws   + (size_t)Bn * (Qn + 1);

    // 8192 tiles of 16q x 32g, 8 waves (256 thr) per block -> 1024 blocks
    build_cost_wmma<<<dim3((Bn * (Qn / 16)) / 8), dim3(256), 0, stream>>>(
        sem, obj, cen, gio, lab, costT);

    hungarian_kernel<<<dim3(Bn), dim3(256), 0, stream>>>(
        costT, nact, v_ws, minv_ws, way_ws, p_ws, used_ws, out);
}
